// SpectralBranch_HIMdown_14370960573107
// MI455X (gfx1250) — matrix operand order repaired and checked by reading, not compiled, not run on hardware
//
#include <hip/hip_runtime.h>
#include <hip/hip_bf16.h>

// ---------------------------------------------------------------------------
// MI455X (gfx1250, wave32) implementation.
//   K1: kv GEMM + bias; L2-normalize k and pre-scale by temp*log2e; emit
//       khat (f16, [n][c]) and v (f16, [c][n'], key-interleaved) per (b,h).
//   K2: flash attention, V_WMMA_F32_16X16X32_F16, DPP16-fused softmax
//       reductions, TDM (tensor_load_to_lds) K/V staging + s_wait_tensorcnt.
//   K3: fused depthwise transposed-conv (stride2, 4x4) + pointwise 64x128.
// ---------------------------------------------------------------------------

typedef __attribute__((ext_vector_type(16))) _Float16 v16h;
typedef __attribute__((ext_vector_type(2)))  _Float16 h2v;
typedef __attribute__((ext_vector_type(8)))  float    v8f;
typedef __attribute__((ext_vector_type(4)))  float    f4;
typedef unsigned int u32x4 __attribute__((ext_vector_type(4)));
typedef int          i32x8 __attribute__((ext_vector_type(8)));
typedef int          i32x4 __attribute__((ext_vector_type(4)));

union Frag16 { v16h v; f4 q[2]; };

#define NB     2
#define NHEAD  8
#define CH     16      // channels per head
#define NKEY   512
#define NQ     16384   // 128*128 queries per (b,h)
#define CDIM   256     // codebook feature dim == W_qkv inner dim
#define LOG2E  1.4426950408889634f

#if __has_builtin(__builtin_amdgcn_exp2f)
#define EXP2(x) __builtin_amdgcn_exp2f(x)
#else
#define EXP2(x) __expf((x) * 0.6931471805599453f)
#endif

#if __has_builtin(__builtin_amdgcn_tensor_load_to_lds) && \
    __has_builtin(__builtin_amdgcn_s_wait_tensorcnt)
#define USE_TDM 1
#endif

// ---- raw single-instruction max (avoids maxnum canonicalize ops) ----------
static __device__ __forceinline__ float fmax_raw(float a, float b) {
    float d;
    asm("v_max_num_f32 %0, %1, %2" : "=v"(d) : "v"(a), "v"(b));
    return d;
}

// ---- 16-lane (DPP row) all-reduce: 4 DPP-fused VALU ops, no LDS -----------
static __device__ __forceinline__ float redmax16(float x) {
    asm("v_max_num_f32_dpp %0, %1, %2 quad_perm:[1,0,3,2] row_mask:0xf bank_mask:0xf bound_ctrl:1"
        : "=v"(x) : "v"(x), "v"(x));
    asm("v_max_num_f32_dpp %0, %1, %2 quad_perm:[2,3,0,1] row_mask:0xf bank_mask:0xf bound_ctrl:1"
        : "=v"(x) : "v"(x), "v"(x));
    asm("v_max_num_f32_dpp %0, %1, %2 row_half_mirror row_mask:0xf bank_mask:0xf bound_ctrl:1"
        : "=v"(x) : "v"(x), "v"(x));
    asm("v_max_num_f32_dpp %0, %1, %2 row_mirror row_mask:0xf bank_mask:0xf bound_ctrl:1"
        : "=v"(x) : "v"(x), "v"(x));
    return x;
}
static __device__ __forceinline__ float redsum16(float x) {
    asm("v_add_f32_dpp %0, %1, %2 quad_perm:[1,0,3,2] row_mask:0xf bank_mask:0xf bound_ctrl:1"
        : "=v"(x) : "v"(x), "v"(x));
    asm("v_add_f32_dpp %0, %1, %2 quad_perm:[2,3,0,1] row_mask:0xf bank_mask:0xf bound_ctrl:1"
        : "=v"(x) : "v"(x), "v"(x));
    asm("v_add_f32_dpp %0, %1, %2 row_half_mirror row_mask:0xf bank_mask:0xf bound_ctrl:1"
        : "=v"(x) : "v"(x), "v"(x));
    asm("v_add_f32_dpp %0, %1, %2 row_mirror row_mask:0xf bank_mask:0xf bound_ctrl:1"
        : "=v"(x) : "v"(x), "v"(x));
    return x;
}

// ---- packed f32x2 -> f16x2 conversion (one v_cvt_pk_rtz_f16_f32) ----------
static __device__ __forceinline__ h2v pack_f16(float a, float b) {
#if __has_builtin(__builtin_amdgcn_cvt_pkrtz)
    return __builtin_bit_cast(h2v, __builtin_amdgcn_cvt_pkrtz(a, b));
#else
    return (h2v){(_Float16)a, (_Float16)b};
#endif
}

static __device__ __forceinline__ float rcp_fast(float x) {
#if __has_builtin(__builtin_amdgcn_rcpf)
    return __builtin_amdgcn_rcpf(x);
#else
    return 1.0f / x;
#endif
}

#ifdef USE_TDM
// ---- Tensor Data Mover: 1-D global->LDS copy (8-byte units) ---------------
// D# per CDNA5 ISA ch.8: group0 = {count=1, lds_addr, global_addr(57b),
// type=2}; group1 = {data_size=3, tensor_dim0=tile_dim0=nbytes/8,
// tensor_dim1=tile_dim1=1, dim0_stride=nbytes/8}; groups2/3 zero (<=2D).
// This toolchain exposes the 6-arg builtin form:
//   (u32x4 g0, i32x8 g1, i32x4 g2, i32x4 g3, i32x8 extra, i32 cpol)
static __device__ __forceinline__ void tdm_copy_to_lds(const void* gptr,
                                                       unsigned lds_off,
                                                       unsigned nbytes)
{
    const unsigned long long ga = (unsigned long long)gptr;
    const unsigned n8 = nbytes >> 3;              // 8-byte elements (<=65535)
    u32x4 g0;
    g0[0] = 1u;                                   // count=1, user descriptor
    g0[1] = lds_off;                              // LDS byte address
    g0[2] = (unsigned)ga;                         // global_addr[31:0]
    g0[3] = (unsigned)((ga >> 32) & 0x1FFFFFFu)   // global_addr[56:32]
          | (2u << 30);                           // type = 2 ("image")
    i32x8 g1;
    g1[0] = (int)(3u << 16);                      // data_size = 3 (8 bytes)
    g1[1] = (int)(n8 << 16);                      // tensor_dim0 @ bits 79:48
    g1[2] = (int)(1u << 16);                      // tensor_dim1 = 1 @ 111:80
    g1[3] = (int)(n8 << 16);                      // tile_dim0  @ bits 127:112
    g1[4] = 1;                                    // tile_dim1 = 1
    g1[5] = (int)n8;                              // tensor_dim0_stride[31:0]
    g1[6] = 0;
    g1[7] = 0;
    const i32x4 z4 = {0, 0, 0, 0};
    const i32x8 z8 = {0, 0, 0, 0, 0, 0, 0, 0};
    __builtin_amdgcn_tensor_load_to_lds(g0, g1, z4, z4, z8, 0);
}
#endif

// ---------------------------------------------------------------------------
// K1: kv = codebook @ W_qkv^T + b_qkv.
//   k half: L2-normalize per (head,key), multiply by temperature[h]*log2e,
//           store khat[b][h][n][c] (f16).
//   v half: store v[b][h][c][n'] (f16) with keys interleaved inside each
//           32-key group: n' = (n/32)*32 + 2*(n%16) + ((n%32)/16), matching
//           the packed-P K ordering used by the attention kernel.
// ---------------------------------------------------------------------------
__global__ __launch_bounds__(256)
void kv_prep_kernel(const float* __restrict__ codebook,
                    const float* __restrict__ Wqkv,
                    const float* __restrict__ bqkv,
                    const float* __restrict__ temperature,
                    _Float16* __restrict__ khat,
                    _Float16* __restrict__ vhT)
{
    __shared__ float cb_s[CDIM];
    const int row = blockIdx.x;            // b*512 + n
    const int b = row >> 9, n = row & 511;
    const int j = threadIdx.x;             // output channel 0..255

    cb_s[j] = codebook[(size_t)row * CDIM + j];
    __syncthreads();

    const f4* wr  = (const f4*)(Wqkv + (size_t)j * CDIM);
    const f4* cb4 = (const f4*)cb_s;
    float acc = bqkv[j];
#pragma unroll 8
    for (int t = 0; t < CDIM / 4; ++t) {
        f4 wv = wr[t], cv = cb4[t];
        acc += wv[0]*cv[0] + wv[1]*cv[1] + wv[2]*cv[2] + wv[3]*cv[3];
    }

    if (j < 128) {
        float ss = acc * acc;               // 16-lane group == one head
        ss = redsum16(ss);
        float norm = fmaxf(sqrtf(ss), 1e-12f);
        const int h = j >> 4, c = j & 15;
        float kn = (acc / norm) * (temperature[h] * LOG2E);
        khat[(((size_t)b * NHEAD + h) * NKEY + n) * CH + c] = (_Float16)kn;
    } else {
        const int j2 = j - 128;
        const int h = j2 >> 4, c = j2 & 15;
        const int grp = n >> 5, within = n & 31;
        const int np = (grp << 5) | (((within & 15) << 1) | (within >> 4));
        vhT[(((size_t)b * NHEAD + h) * CH + c) * NKEY + np] = (_Float16)acc;
    }
}

// ---------------------------------------------------------------------------
// K2: flash attention.  grid = (64 q-blocks, 16 b*h), 256 threads (8 waves).
// Each wave owns TWO 16-query tiles (amortizes K/V staging + B fragments).
// Per 32-key iteration and per tile:
//   2x QK^T WMMA (K padded 16->32 via zero LDS rows)
//   -> online softmax (DPP-fused max-reduce, deferred sum reduce)
//   -> packed-P LDS transpose -> 1x PV WMMA with full K=32.
// K/V staging uses the Tensor Data Mover (one wave issues 2 descriptors,
// waits on TENSORcnt, then the workgroup barrier publishes the data).
// ---------------------------------------------------------------------------
__global__ __launch_bounds__(256)
void flash_attn_kernel(const float* __restrict__ q,
                       const _Float16* __restrict__ khat,
                       const _Float16* __restrict__ vhT,
                       float* __restrict__ attn_out)
{
    __shared__ _Float16 khat_s[(NKEY + 16) * CH];  // 16.5 KB [n][c] + 16 zero rows
    __shared__ _Float16 vhT_s [CH * NKEY];         // 16 KB   [c][n'] (interleaved)
    __shared__ _Float16 q_s   [8][2][16 * 16];     //  8 KB   per-wave q tiles [m][c]
    __shared__ _Float16 p_s   [8][16 * 32];        //  8 KB   per-wave P tile [m][k']

    const int bh   = blockIdx.y;                   // 0..15
    const int b    = bh >> 3, h = bh & 7;
    const int tid  = threadIdx.x;
    const int w    = tid >> 5;                     // wave id 0..7
    const int lane = tid & 31;
    const int half = lane >> 4;                    // 0: lanes 0-15, 1: 16-31
    const int mrow = lane & 15;

    // ---- stage khat / vhT for this (b,h) into LDS
#ifdef USE_TDM
    if (w == 0) {   // TDM ignores EXEC and issues per-wave: exactly one wave
        tdm_copy_to_lds(khat + (size_t)bh * NKEY * CH,
                        (unsigned)(size_t)&khat_s[0], NKEY * CH * 2);
        tdm_copy_to_lds(vhT + (size_t)bh * CH * NKEY,
                        (unsigned)(size_t)&vhT_s[0], CH * NKEY * 2);
        __builtin_amdgcn_s_wait_tensorcnt((short)0);
    }
    if (tid < 32) ((f4*)khat_s)[NKEY * CH / 8 + tid] = (f4){0.f, 0.f, 0.f, 0.f};
#else
    {
        const f4* sk = (const f4*)(khat + (size_t)bh * NKEY * CH);
        f4* dk = (f4*)khat_s;
        for (int i = tid; i < NKEY * CH / 8; i += 256) dk[i] = sk[i];
        if (tid < 32) dk[NKEY * CH / 8 + tid] = (f4){0.f, 0.f, 0.f, 0.f};
        const f4* sv = (const f4*)(vhT + (size_t)bh * CH * NKEY);
        f4* dv = (f4*)vhT_s;
        for (int i = tid; i < CH * NKEY / 8; i += 256) dv[i] = sv[i];
    }
#endif

    // ---- stage this wave's two 16x16 q tiles (channel-major -> [m][c])
    const int qbase0 = blockIdx.x * 256 + w * 16;
#pragma unroll
    for (int t = 0; t < 2; ++t) {
        const float* qp = q + ((size_t)b * 128 + h * CH) * NQ + qbase0 + t * 128;
        for (int i = lane; i < 256; i += 32) {
            const int m = i & 15, c = i >> 4;
            q_s[w][t][m * 16 + c] = (_Float16)qp[(size_t)c * NQ + m];
        }
    }
    __syncthreads();

    // ---- A fragments for Q (16x32, K=16..31 zero padding)
    Frag16 aq[2];
#pragma unroll
    for (int t = 0; t < 2; ++t) {
        aq[t].q[0] = *(const f4*)&q_s[w][t][mrow * 16];
        aq[t].q[1] = *(const f4*)&q_s[w][t][mrow * 16 + 8];
    }

    // khat B-fragment pointers: lanes 16-31 (K rows 16..31) always read the
    // zero-pad rows; no per-iteration cndmask needed.
    const _Float16* kb  = &khat_s[(half ? (NKEY + mrow) : mrow) * CH];
    const int kstep = half ? 0 : 32 * CH;      // advance per 32-key group
    const int koff1 = half ? 0 : 16 * CH;      // second 16-key tile

    const v8f zero8 = {0.f,0.f,0.f,0.f,0.f,0.f,0.f,0.f};
    v8f acc[2] = {zero8, zero8};
    float mrun[2][8], lrun[2][8];
#pragma unroll
    for (int t = 0; t < 2; ++t)
#pragma unroll
        for (int r = 0; r < 8; ++r) { mrun[t][r] = -1e30f; lrun[t][r] = 0.f; }

    for (int kt = 0; kt < NKEY / 32; ++kt) {
        // ---- shared B fragments for this 32-key group
        Frag16 bk0, bk1, bv;
        {
            const f4* r0 = (const f4*)(kb + kt * kstep);
            const f4* r1 = (const f4*)(kb + kt * kstep + koff1);
            bk0.q[0] = r0[0]; bk0.q[1] = r0[1];
            bk1.q[0] = r1[0]; bk1.q[1] = r1[1];
            const f4* vr = (const f4*)&vhT_s[mrow * NKEY + kt * 32 + half * 16];
            bv.q[0] = vr[0]; bv.q[1] = vr[1];
        }

#pragma unroll
        for (int t = 0; t < 2; ++t) {
            v8f s0 = __builtin_amdgcn_wmma_f32_16x16x32_f16(
                false, aq[t].v, false, bk0.v, (short)0, zero8, false, false);
            v8f s1 = __builtin_amdgcn_wmma_f32_16x16x32_f16(
                false, aq[t].v, false, bk1.v, (short)0, zero8, false, false);

            // ---- online softmax (scores already in exp2 domain)
            float p0[8], p1[8];
#pragma unroll
            for (int r = 0; r < 8; ++r) {
                float red = redmax16(fmax_raw(s0[r], s1[r]));
                const float mnew = fmax_raw(mrun[t][r], red);
                const float fs = EXP2(mrun[t][r] - mnew);
                p0[r] = EXP2(s0[r] - mnew);
                p1[r] = EXP2(s1[r] - mnew);
                lrun[t][r] = lrun[t][r] * fs + (p0[r] + p1[r]); // per-lane partial
                mrun[t][r] = mnew;
                acc[t][r] *= fs;
            }

            // ---- D-layout -> A-layout transpose of P (packed b32 stores);
            //      K' ordering = 2*col + tile matches vhT interleave.
#pragma unroll
            for (int r = 0; r < 8; ++r) {
                const int m = r + half * 8;
                *(h2v*)&p_s[w][m * 32 + 2 * mrow] = pack_f16(p0[r], p1[r]);
            }
            Frag16 pa;
            pa.q[0] = *(const f4*)&p_s[w][mrow * 32 + half * 16];
            pa.q[1] = *(const f4*)&p_s[w][mrow * 32 + half * 16 + 8];

            acc[t] = __builtin_amdgcn_wmma_f32_16x16x32_f16(
                false, pa.v, false, bv.v, (short)0, acc[t], false, false);
        }
    }

    // ---- finalize: reduce deferred row sums once, scale by 1/sum, store
#pragma unroll
    for (int t = 0; t < 2; ++t) {
#pragma unroll
        for (int r = 0; r < 8; ++r) {
            const float rs = redsum16(lrun[t][r]);
            acc[t][r] = acc[t][r] * rcp_fast(rs);
        }
        float* op = attn_out + ((size_t)b * 128 + h * CH + mrow) * NQ
                             + qbase0 + t * 128 + half * 8;
        *(f4*)op       = (f4){acc[t][0], acc[t][1], acc[t][2], acc[t][3]};
        *(f4*)(op + 4) = (f4){acc[t][4], acc[t][5], acc[t][6], acc[t][7]};
    }
}

// ---------------------------------------------------------------------------
// K3: fused depthwise transposed conv (stride 2, 4x4, per-channel) + 64x128
// pointwise.  grid = NB * 256 blocks; each block: 16x16 output tile, 256 thr,
// channels processed in 4 chunks of 32 through LDS.
// ---------------------------------------------------------------------------
__global__ __launch_bounds__(256)
void deconv_pw_kernel(const float* __restrict__ ain,   // (2,128,128,128)
                      const float* __restrict__ dwk,   // (128,1,4,4)
                      const float* __restrict__ pw,    // (64,128)
                      float* __restrict__ out)         // (2,64,256,256)
{
    __shared__ float in_s [32 * 100];   // 12.5 KB : 32 ch x 10x10 input halo
    __shared__ float pwT_s[32 * 64];    //  8   KB : pw transposed [i][o]
    __shared__ float dwk_s[32 * 16];    //  2   KB

    const int blk   = blockIdx.x;
    const int bb    = blk >> 8;
    const int t     = blk & 255;
    const int tileY = t >> 4, tileX = t & 15;
    const int Y0 = tileY * 16, X0 = tileX * 16;
    const int tid = threadIdx.x;
    const int ty = tid >> 4, tx = tid & 15;
    const int Y = Y0 + ty, X = X0 + tx;

    // transposed-conv taps: ky in {Y&1, (Y&1)+2}, same for kx; weight is
    // dw_kernel[3-ky][3-kx]; input (y,x) with y=(Y+ky-2)/2.
    const int py = Y & 1, px = X & 1;
    const int y0base = Y0 / 2 - 1, x0base = X0 / 2 - 1;
    const int yy0 = (Y + py - 2) / 2 - y0base, yy1 = yy0 + 1;
    const int xx0 = (X + px - 2) / 2 - x0base, xx1 = xx0 + 1;
    const int wi00 = (3 - py) * 4 + (3 - px);
    const int wi01 = (3 - py) * 4 + (1 - px);
    const int wi10 = (1 - py) * 4 + (3 - px);
    const int wi11 = (1 - py) * 4 + (1 - px);

    f4 acc4[16];
#pragma unroll
    for (int o = 0; o < 16; ++o) acc4[o] = (f4){0.f, 0.f, 0.f, 0.f};

    for (int ch0 = 0; ch0 < 128; ch0 += 32) {
        __syncthreads();
        // load input halo (zero-padded), pw chunk (transposed), dw chunk
        for (int e = tid; e < 3200; e += 256) {
            const int i = e / 100, rem = e % 100;
            const int yy = rem / 10, xx = rem % 10;
            const int gy = y0base + yy, gx = x0base + xx;
            float val = 0.f;
            if ((unsigned)gy < 128u && (unsigned)gx < 128u)
                val = ain[(((size_t)bb * 128 + ch0 + i) * 128 + gy) * 128 + gx];
            in_s[e] = val;
        }
        for (int e = tid; e < 2048; e += 256) {
            const int i = e >> 6, o = e & 63;
            pwT_s[e] = pw[o * 128 + ch0 + i];
        }
        for (int e = tid; e < 512; e += 256) {
            const int i = e >> 4, k = e & 15;
            dwk_s[e] = dwk[(ch0 + i) * 16 + k];
        }
        __syncthreads();

        for (int i = 0; i < 32; ++i) {
            const float* cin = &in_s[i * 100];
            const float* cw  = &dwk_s[i * 16];
            const float tmp = cw[wi00] * cin[yy0 * 10 + xx0]
                            + cw[wi01] * cin[yy0 * 10 + xx1]
                            + cw[wi10] * cin[yy1 * 10 + xx0]
                            + cw[wi11] * cin[yy1 * 10 + xx1];
            const f4* pr = (const f4*)&pwT_s[i * 64];
#pragma unroll
            for (int o = 0; o < 16; ++o) acc4[o] += pr[o] * tmp;
        }
    }

    float* op = out + (((size_t)bb * 64) * 256 + Y) * 256 + X;
#pragma unroll
    for (int o = 0; o < 16; ++o) {
        op[(size_t)(o * 4 + 0) * 65536] = acc4[o][0];
        op[(size_t)(o * 4 + 1) * 65536] = acc4[o][1];
        op[(size_t)(o * 4 + 2) * 65536] = acc4[o][2];
        op[(size_t)(o * 4 + 3) * 65536] = acc4[o][3];
    }
}

// ---------------------------------------------------------------------------
extern "C" void kernel_launch(void* const* d_in, const int* in_sizes, int n_in,
                              void* d_out, int out_size, void* d_ws, size_t ws_size,
                              hipStream_t stream) {
    // setup_inputs order: x, q, codebook, W_qkv, b_qkv, temperature, dw_kernel, pw_kernel
    const float* q        = (const float*)d_in[1];
    const float* codebook = (const float*)d_in[2];
    const float* Wqkv     = (const float*)d_in[3];
    const float* bqkv     = (const float*)d_in[4];
    const float* temp     = (const float*)d_in[5];
    const float* dwk      = (const float*)d_in[6];
    const float* pw       = (const float*)d_in[7];
    float* out = (float*)d_out;

    char* ws = (char*)d_ws;
    _Float16* khat = (_Float16*)ws;                       // 256 KB
    _Float16* vhT  = (_Float16*)(ws + 262144);            // 256 KB
    float*    attn = (float*)(ws + (1 << 20));            // 16 MB

    kv_prep_kernel<<<NB * NKEY, 256, 0, stream>>>(codebook, Wqkv, bqkv, temp,
                                                  khat, vhT);

    dim3 g2(NQ / 256, NB * NHEAD);
    flash_attn_kernel<<<g2, 256, 0, stream>>>(q, khat, vhT, attn);

    deconv_pw_kernel<<<NB * 256, 256, 0, stream>>>(attn, dwk, pw, out);
}
